// Pointnet_FP_9792525435024
// MI455X (gfx1250) — matrix operand order repaired and checked by reading, not compiled, not run on hardware
//
#include <hip/hip_runtime.h>
#include <cfloat>

// ---------------- types ----------------
typedef __attribute__((ext_vector_type(16))) __bf16 v16bf;
typedef __attribute__((ext_vector_type(8)))  float  v8f;

union Frag { uint4 u[2]; v16bf v; };

// ---------------- problem constants ----------------
#define BB    8
#define NP    4096
#define MP    1024
#define C1c   128
#define C2c   256
#define K1    384          // C2 + C1
#define NF    256          // mlp widths (both layers)
#define ROWS  (BB*NP)      // 32768
#define GEMM_BLOCKS (ROWS/128)   // 256
#define BN_EPS 1e-3f

__device__ __forceinline__ unsigned short f2bf(float f) {
    unsigned u = __builtin_bit_cast(unsigned, f);
    u += 0x7fffu + ((u >> 16) & 1u);          // round-to-nearest-even
    return (unsigned short)(u >> 16);
}

// ---------------- weight transpose + bf16 convert:  w[K][N] f32 -> wt[N][K] bf16 ----------------
__global__ void k_wt(const float* __restrict__ w, unsigned short* __restrict__ wt, int K, int N) {
    int id = blockIdx.x * 256 + threadIdx.x;
    if (id >= K * N) return;
    int n = id / K, k = id % K;
    wt[id] = f2bf(w[(size_t)k * N + n]);
}

// ---------------- three_nn + IDW interpolation + concat -> A1 bf16 [ROWS][384] ----------------
__global__ __launch_bounds__(256) void k_interp(const float* __restrict__ xyz1,
                                                const float* __restrict__ xyz2,
                                                const float* __restrict__ points1,
                                                const float* __restrict__ points2,
                                                unsigned short* __restrict__ A1) {
    __shared__ float sx[MP], sy[MP], sz[MP];
    __shared__ int   sidx[256][3];
    __shared__ float swt[256][3];
    const int b = blockIdx.y;
    const int tid = threadIdx.x;

    const float* x2 = xyz2 + (size_t)b * MP * 3;
    for (int i = tid; i < MP; i += 256) {
        sx[i] = x2[i * 3 + 0];
        sy[i] = x2[i * 3 + 1];
        sz[i] = x2[i * 3 + 2];
    }
    __syncthreads();

    const int n = blockIdx.x * 256 + tid;
    const float* p = xyz1 + ((size_t)b * NP + n) * 3;
    const float px = p[0], py = p[1], pz = p[2];

    float d0 = FLT_MAX, d1 = FLT_MAX, d2 = FLT_MAX;
    int   i0 = 0, i1 = 0, i2 = 0;
#pragma unroll 4
    for (int m = 0; m < MP; ++m) {
        float dx = px - sx[m], dy = py - sy[m], dz = pz - sz[m];
        float d = dx * dx + dy * dy + dz * dz;
        if (d < d2) {
            if (d < d1) {
                if (d < d0) { d2 = d1; i2 = i1; d1 = d0; i1 = i0; d0 = d; i0 = m; }
                else        { d2 = d1; i2 = i1; d1 = d;  i1 = m; }
            } else          { d2 = d;  i2 = m; }
        }
    }
    d0 = fmaxf(d0, 1e-10f); d1 = fmaxf(d1, 1e-10f); d2 = fmaxf(d2, 1e-10f);
    float w0 = 1.f / d0, w1 = 1.f / d1, w2 = 1.f / d2;
    float s = 1.f / (w0 + w1 + w2);
    sidx[tid][0] = i0; sidx[tid][1] = i1; sidx[tid][2] = i2;
    swt[tid][0] = w0 * s; swt[tid][1] = w1 * s; swt[tid][2] = w2 * s;
    __syncthreads();

    // block-cooperative coalesced gather: one row at a time, 256 channels wide
    const float* P2 = points2 + (size_t)b * MP * C2c;
    const int c = tid;
    for (int r = 0; r < 256; ++r) {
        const int nr = blockIdx.x * 256 + r;
        const size_t row = (size_t)b * NP + nr;
        const int j0 = sidx[r][0], j1 = sidx[r][1], j2 = sidx[r][2];
        const float a0 = swt[r][0], a1 = swt[r][1], a2 = swt[r][2];
        float v = a0 * P2[(size_t)j0 * C2c + c]
                + a1 * P2[(size_t)j1 * C2c + c]
                + a2 * P2[(size_t)j2 * C2c + c];
        A1[row * K1 + c] = f2bf(v);
        if (c < C1c)
            A1[row * K1 + C2c + c] = f2bf(points1[row * C1c + c]);
    }
}

// ---------------- GEMM (bf16 WMMA, f32 acc) + per-block deterministic BN partial sums ----------
// A [ROWS][K] bf16 row-major, Wt [NF][K] bf16 (W transposed), Y [ROWS][NF] f32.
// Block = 256 threads = 8 waves; block tile = 128M x 256N; wave tile = 32M x 128N.
__global__ __launch_bounds__(256) void k_gemm(const unsigned short* __restrict__ A,
                                              const unsigned short* __restrict__ Wt,
                                              float* __restrict__ Y,
                                              float* __restrict__ psum,
                                              float* __restrict__ psq,
                                              int K) {
    __shared__ float lsum[NF][8];
    __shared__ float lsq[NF][8];

    const int tid  = threadIdx.x;
    const int wave = tid >> 5;
    const int lane = tid & 31;
    const int wm   = wave & 3;          // 4 waves along M
    const int wn   = wave >> 2;         // 2 waves along N
    const int mbase = blockIdx.x * 128 + wm * 32;
    const int nbase = wn * 128;
    const int lm  = lane & 15;
    const int hi  = lane >> 4;          // lane half
    const int kgA = hi << 3;            // A: K chunks at kgA, kgA+16 (ISA 16-bit A layout)
    const int kgB = hi << 3;            // B: K chunks at kgB, kgB+16 (same as A)

    v8f acc[2][8];
    const v8f vzero = {0.f, 0.f, 0.f, 0.f, 0.f, 0.f, 0.f, 0.f};
#pragma unroll
    for (int i = 0; i < 2; ++i)
#pragma unroll
        for (int j = 0; j < 8; ++j) acc[i][j] = vzero;

    const unsigned short* a0p = A + (size_t)(mbase + lm) * K + kgA;
    const unsigned short* a1p = A + (size_t)(mbase + 16 + lm) * K + kgA;

    for (int kb = 0; kb < K; kb += 32) {
        __builtin_prefetch(a0p + kb + 32, 0, 1);
        __builtin_prefetch(a1p + kb + 32, 0, 1);
        Frag fa0, fa1;
        fa0.u[0] = *(const uint4*)(a0p + kb);
        fa0.u[1] = *(const uint4*)(a0p + kb + 16);
        fa1.u[0] = *(const uint4*)(a1p + kb);
        fa1.u[1] = *(const uint4*)(a1p + kb + 16);
#pragma unroll
        for (int t = 0; t < 8; ++t) {
            const unsigned short* bp = Wt + (size_t)(nbase + t * 16 + lm) * K + kb + kgB;
            Frag fb;
            fb.u[0] = *(const uint4*)(bp);
            fb.u[1] = *(const uint4*)(bp + 16);
            acc[0][t] = __builtin_amdgcn_wmma_f32_16x16x32_bf16(
                false, fa0.v, false, fb.v, (short)0, acc[0][t], false, false);
            acc[1][t] = __builtin_amdgcn_wmma_f32_16x16x32_bf16(
                false, fa1.v, false, fb.v, (short)0, acc[1][t], false, false);
        }
    }

    // store Y + stage per-thread BN partials in fixed LDS slots (deterministic)
    const int slot = wm * 2 + hi;       // 8 contributors per column
#pragma unroll
    for (int t = 0; t < 8; ++t) {
        const int col = nbase + t * 16 + lm;
        float s = 0.f, q = 0.f;
#pragma unroll
        for (int mi = 0; mi < 2; ++mi) {
            const int row0 = mbase + mi * 16 + (hi << 3);
#pragma unroll
            for (int r = 0; r < 8; ++r) {
                float v = acc[mi][t][r];
                Y[(size_t)(row0 + r) * NF + col] = v;
                s += v;
                q += v * v;
            }
        }
        lsum[col][slot] = s;
        lsq[col][slot]  = q;
    }
    __syncthreads();
    // fixed-order per-column reduce -> per-block partials
    {
        const int col = tid;            // 256 threads == NF
        float s = 0.f, q = 0.f;
#pragma unroll
        for (int j = 0; j < 8; ++j) { s += lsum[col][j]; q += lsq[col][j]; }
        psum[(size_t)blockIdx.x * NF + col] = s;
        psq [(size_t)blockIdx.x * NF + col] = q;
    }
}

// ---------------- deterministic reduce of per-block partials ----------------
__global__ void k_reduce(const float* __restrict__ ps, const float* __restrict__ pq,
                         float* __restrict__ gs, float* __restrict__ gq, int nblocks) {
    const int f = threadIdx.x;
    float s = 0.f, q = 0.f;
    for (int b = 0; b < nblocks; ++b) {
        s += ps[(size_t)b * NF + f];
        q += pq[(size_t)b * NF + f];
    }
    gs[f] = s;
    gq[f] = q;
}

// ---------------- BN (training stats) + ReLU, write bf16 activations ----------------
__global__ void k_bn_bf16(const float* __restrict__ Y,
                          const float* __restrict__ gs, const float* __restrict__ gq,
                          const float* __restrict__ gamma, const float* __restrict__ beta,
                          unsigned short* __restrict__ out) {
    const size_t e = ((size_t)blockIdx.x * 256 + threadIdx.x) * 4;
    const int f = (int)(e & (NF - 1));
    const float inv_cnt = 1.0f / (float)ROWS;
    float4 y = *(const float4*)(Y + e);
    float yv[4] = {y.x, y.y, y.z, y.w};
    unsigned short o16[4];
#pragma unroll
    for (int j = 0; j < 4; ++j) {
        const int ff = f + j;
        const float mean = gs[ff] * inv_cnt;
        const float var  = gq[ff] * inv_cnt - mean * mean;
        float v = (yv[j] - mean) * rsqrtf(var + BN_EPS) * gamma[ff] + beta[ff];
        o16[j] = f2bf(fmaxf(v, 0.f));
    }
    uint2 pack;
    pack.x = (unsigned)o16[0] | ((unsigned)o16[1] << 16);
    pack.y = (unsigned)o16[2] | ((unsigned)o16[3] << 16);
    *(uint2*)(out + e) = pack;
}

// ---------------- BN + ReLU, write f32 final output ----------------
__global__ void k_bn_f32(const float* __restrict__ Y,
                         const float* __restrict__ gs, const float* __restrict__ gq,
                         const float* __restrict__ gamma, const float* __restrict__ beta,
                         float* __restrict__ out) {
    const size_t e = ((size_t)blockIdx.x * 256 + threadIdx.x) * 4;
    const int f = (int)(e & (NF - 1));
    const float inv_cnt = 1.0f / (float)ROWS;
    float4 y = *(const float4*)(Y + e);
    float yv[4] = {y.x, y.y, y.z, y.w};
    float4 o;
    float* ov = (float*)&o;
#pragma unroll
    for (int j = 0; j < 4; ++j) {
        const int ff = f + j;
        const float mean = gs[ff] * inv_cnt;
        const float var  = gq[ff] * inv_cnt - mean * mean;
        float v = (yv[j] - mean) * rsqrtf(var + BN_EPS) * gamma[ff] + beta[ff];
        ov[j] = fmaxf(v, 0.f);
    }
    *(float4*)(out + e) = o;
}

// ---------------- host side ----------------
extern "C" void kernel_launch(void* const* d_in, const int* in_sizes, int n_in,
                              void* d_out, int out_size, void* d_ws, size_t ws_size,
                              hipStream_t stream) {
    (void)in_sizes; (void)n_in; (void)out_size; (void)ws_size;
    const float* xyz1    = (const float*)d_in[0];
    const float* xyz2    = (const float*)d_in[1];
    const float* points1 = (const float*)d_in[2];
    const float* points2 = (const float*)d_in[3];
    const float* w1      = (const float*)d_in[4];
    const float* gamma1  = (const float*)d_in[5];
    const float* beta1   = (const float*)d_in[6];
    const float* w2      = (const float*)d_in[7];
    const float* gamma2  = (const float*)d_in[8];
    const float* beta2   = (const float*)d_in[9];
    float* out = (float*)d_out;

    char* ws = (char*)d_ws;
    size_t off = 0;
    auto take = [&](size_t bytes) { size_t o = off; off = (off + bytes + 255) & ~(size_t)255; return o; };

    // region0 aliased: A1 (live interp->gemm1) and Y2 (live gemm2->bn2) — disjoint lifetimes
    const size_t region0 = take((size_t)ROWS * NF * 4);          // 33.5 MB (>= A1's 25.2 MB)
    unsigned short* A1 = (unsigned short*)(ws + region0);
    float*          Y2 = (float*)(ws + region0);
    float*          Y1 = (float*)(ws + take((size_t)ROWS * NF * 4));
    unsigned short* A2 = (unsigned short*)(ws + take((size_t)ROWS * NF * 2));
    unsigned short* Wt1 = (unsigned short*)(ws + take((size_t)NF * K1 * 2));
    unsigned short* Wt2 = (unsigned short*)(ws + take((size_t)NF * NF * 2));
    float* psum = (float*)(ws + take((size_t)GEMM_BLOCKS * NF * 4));
    float* psq  = (float*)(ws + take((size_t)GEMM_BLOCKS * NF * 4));
    float* gs   = (float*)(ws + take(NF * 4));
    float* gq   = (float*)(ws + take(NF * 4));

    // weights -> transposed bf16
    k_wt<<<(K1 * NF + 255) / 256, 256, 0, stream>>>(w1, Wt1, K1, NF);
    k_wt<<<(NF * NF + 255) / 256, 256, 0, stream>>>(w2, Wt2, NF, NF);

    // three_nn + interpolation + concat -> A1
    k_interp<<<dim3(NP / 256, BB), 256, 0, stream>>>(xyz1, xyz2, points1, points2, A1);

    const int bn_blocks = (int)(((size_t)ROWS * NF / 4) / 256);  // 8192

    // layer 1
    k_gemm  <<<GEMM_BLOCKS, 256, 0, stream>>>(A1, Wt1, Y1, psum, psq, K1);
    k_reduce<<<1, NF, 0, stream>>>(psum, psq, gs, gq, GEMM_BLOCKS);
    k_bn_bf16<<<bn_blocks, 256, 0, stream>>>(Y1, gs, gq, gamma1, beta1, A2);

    // layer 2
    k_gemm  <<<GEMM_BLOCKS, 256, 0, stream>>>(A2, Wt2, Y2, psum, psq, NF);
    k_reduce<<<1, NF, 0, stream>>>(psum, psq, gs, gq, GEMM_BLOCKS);
    k_bn_f32<<<bn_blocks, 256, 0, stream>>>(Y2, gs, gq, gamma2, beta2, out);
}
